// MultiHeadAttention_9577777070314
// MI455X (gfx1250) — matrix operand order repaired and checked by reading, not compiled, not run on hardware
//
#include <hip/hip_runtime.h>
#include <hip/hip_bf16.h>

#define B_     2
#define S_     2048
#define D_     1024
#define H_     16
#define DEPTH_ 64
#define BH_    32

typedef __bf16 bf16_t;
typedef __attribute__((ext_vector_type(16))) __bf16 v16bf;
typedef __attribute__((ext_vector_type(8)))  __bf16 v8bf;
typedef __attribute__((ext_vector_type(8)))  float  v8f;

__device__ __forceinline__ v16bf cat16(v8bf lo, v8bf hi) {
  return __builtin_shufflevector(lo, hi, 0,1,2,3,4,5,6,7,8,9,10,11,12,13,14,15);
}
__device__ __forceinline__ v8f wmma_bf16(v16bf a, v16bf b, v8f c) {
  // D = A(16x32 bf16) * B(32x16 bf16) + C(f32)
  return __builtin_amdgcn_wmma_f32_16x16x32_bf16(false, a, false, b, (short)0, c, false, false);
}
// CDNA5 async copy engine: per-lane 16B global -> LDS, tracked by ASYNCcnt.
__device__ __forceinline__ void async_copy_b128(const void* gaddr, void* lds) {
  const unsigned long long ga = (unsigned long long)gaddr;
  const unsigned la = (unsigned)(uintptr_t)lds;   // LDS aperture: addr[31:0] = offset
  asm volatile("global_load_async_to_lds_b128 %0, %1, off"
               :: "v"(la), "v"(ga) : "memory");
}
__device__ __forceinline__ void async_wait0() {
  asm volatile("s_wait_asynccnt 0" ::: "memory");
}

// ---------------------------------------------------------------------------
// Kernel 1/4: Y = X[4096x1024] * W[1024x1024] + b, with mode-dependent store:
//   MODE 0: Q head-split  -> Qh [bh][s][64]   (bf16, row-major)
//   MODE 1: K head-split  -> Kh [bh][s][64]   (bf16, row-major: QK^T B-op cols)
//   MODE 2: V head-split  -> VhT[bh][64][s]   (bf16, transposed: P@V B-op cols)
//   MODE 3: plain f32 row-major (final output projection)
// Block: 256 thr = 8 waves (2x4), tile 128x128, K-chunks of 32.
// ---------------------------------------------------------------------------
template <int MODE>
__global__ __launch_bounds__(256)
void gemm_proj(const float* __restrict__ X, const float* __restrict__ W,
               const float* __restrict__ bias, void* __restrict__ dst) {
  __shared__ bf16_t As[128][40];   // [row][k], 80B row stride (16B aligned)
  __shared__ bf16_t Bs[128][40];   // [col][k] transposed, same padding
  const int t = threadIdx.x, lane = t & 31, wave = t >> 5;
  const int wm = wave >> 2, wn = wave & 3;         // 2 x 4 wave grid
  const int hf = lane >> 4, l16 = lane & 15;
  const int m0 = blockIdx.y * 128, n0 = blockIdx.x * 128;

  v8f acc[4][2] = {};                              // 64 rows x 32 cols per wave

  for (int kk0 = 0; kk0 < D_; kk0 += 32) {
    if (kk0 + 32 < D_) {                           // prefetch next K-chunk
      const int lin = t;                           // one line per thread
      __builtin_prefetch(&X[(size_t)(m0 + (lin >> 1)) * D_ + kk0 + 32 + (lin & 1) * 16], 0, 0);
      __builtin_prefetch(&W[(size_t)(kk0 + 32 + (lin >> 3)) * D_ + n0 + (lin & 7) * 16], 0, 0);
    }
#pragma unroll
    for (int j = 0; j < 4; ++j) {                  // A tile: 128x32 f32 -> bf16
      const int lin = t + 256 * j;                 // 1024 float4
      const int row = lin >> 3, c4 = (lin & 7) << 2;
      const float4 v = *(const float4*)&X[(size_t)(m0 + row) * D_ + kk0 + c4];
      As[row][c4 + 0] = (bf16_t)v.x; As[row][c4 + 1] = (bf16_t)v.y;
      As[row][c4 + 2] = (bf16_t)v.z; As[row][c4 + 3] = (bf16_t)v.w;
    }
#pragma unroll
    for (int j = 0; j < 4; ++j) {                  // B tile: 32x128, store transposed
      const int lin = t + 256 * j;
      const int kk = lin >> 5, c4 = (lin & 31) << 2;
      const float4 v = *(const float4*)&W[(size_t)(kk0 + kk) * D_ + n0 + c4];
      Bs[c4 + 0][kk] = (bf16_t)v.x; Bs[c4 + 1][kk] = (bf16_t)v.y;
      Bs[c4 + 2][kk] = (bf16_t)v.z; Bs[c4 + 3][kk] = (bf16_t)v.w;
    }
    __syncthreads();

    v16bf a[4], bfr[2];
#pragma unroll
    for (int i = 0; i < 4; ++i) {                  // A frag: lanes 0-15 K{0..7,16..23}
      const int row = wm * 64 + i * 16 + l16;      //         lanes 16-31 +8
      a[i] = cat16(*(const v8bf*)&As[row][8 * hf],
                   *(const v8bf*)&As[row][16 + 8 * hf]);
    }
#pragma unroll
    for (int i = 0; i < 2; ++i) {                  // B frag: K{0..7,16..23} +8*half
      const int col = wn * 32 + i * 16 + l16;
      bfr[i] = cat16(*(const v8bf*)&Bs[col][8 * hf],
                     *(const v8bf*)&Bs[col][16 + 8 * hf]);
    }
#pragma unroll
    for (int i = 0; i < 4; ++i)
#pragma unroll
      for (int j2 = 0; j2 < 2; ++j2)
        acc[i][j2] = wmma_bf16(a[i], bfr[j2], acc[i][j2]);
    __syncthreads();
  }

#pragma unroll
  for (int i = 0; i < 4; ++i) {
#pragma unroll
    for (int j2 = 0; j2 < 2; ++j2) {
      const int gc = n0 + wn * 32 + j2 * 16 + l16;
      const float bv = bias[gc];
#pragma unroll
      for (int r = 0; r < 8; ++r) {
        const int gr = m0 + wm * 64 + i * 16 + 8 * hf + r;
        const float val = acc[i][j2][r] + bv;
        const int bb = gr >> 11, ss = gr & (S_ - 1);
        const int hh = gc >> 6,  dd = gc & 63;
        const int bh = (bb << 4) | hh;
        if (MODE == 0 || MODE == 1)
          ((bf16_t*)dst)[((size_t)bh * S_ + ss) * DEPTH_ + dd] = (bf16_t)val;
        else if (MODE == 2)
          ((bf16_t*)dst)[((size_t)bh * DEPTH_ + dd) * S_ + ss] = (bf16_t)val;
        else
          ((float*)dst)[(size_t)gr * D_ + gc] = val;
      }
    }
  }
}

// ---------------------------------------------------------------------------
// Kernel 2: logits = scale*Qh*Kh^T + mask*(-1e9) -> attn buffer (f32),
// with online per-row (max, sumexp) -> rowstats. 8 waves x 16 q-rows.
// K tiles staged to LDS by the async copy engine (pure 16B copies).
// ---------------------------------------------------------------------------
__global__ __launch_bounds__(256)
void attn_logits(const bf16_t* __restrict__ Qh, const bf16_t* __restrict__ Kh,
                 const float* __restrict__ mask, float* __restrict__ attn,
                 float* __restrict__ rowstats) {
  __shared__ bf16_t Ks[128][72];                   // [s_k col][depth], 144B stride
  const int t = threadIdx.x, lane = t & 31, wave = t >> 5;
  const int hf = lane >> 4, l16 = lane & 15;
  const int bh = blockIdx.y, bidx = bh >> 4;
  const int sbase = blockIdx.x * 128 + wave * 16;

  // Q fragments for this wave's 16 rows (kept in registers for whole kernel)
  const bf16_t* qrow = &Qh[((size_t)bh * S_ + sbase + l16) * DEPTH_];
  const v16bf aq0 = cat16(*(const v8bf*)&qrow[8 * hf],      *(const v8bf*)&qrow[16 + 8 * hf]);
  const v16bf aq1 = cat16(*(const v8bf*)&qrow[32 + 8 * hf], *(const v8bf*)&qrow[48 + 8 * hf]);

  float m[8], l[8];
#pragma unroll
  for (int r = 0; r < 8; ++r) { m[r] = -3.0e38f; l[r] = 0.f; }
  const float scale = 0.125f;                      // 1/sqrt(64)

  for (int nc = 0; nc < S_ / 128; ++nc) {
    const int n0 = nc * 128;
#pragma unroll
    for (int j = 0; j < 4; ++j) {                  // Kh rows n0..n0+127, 64 deep
      const int c = t + 256 * j;                   // 1024 x 8-elem chunks
      const int row = c >> 3, o8 = (c & 7) << 3;
      async_copy_b128(&Kh[((size_t)bh * S_ + n0 + row) * DEPTH_ + o8], &Ks[row][o8]);
    }
    async_wait0();
    __syncthreads();
#pragma unroll
    for (int sub = 0; sub < 8; ++sub) {
      const int col = sub * 16 + l16;
      const v16bf b0 = cat16(*(const v8bf*)&Ks[col][8 * hf],
                             *(const v8bf*)&Ks[col][16 + 8 * hf]);
      const v16bf b1 = cat16(*(const v8bf*)&Ks[col][32 + 8 * hf],
                             *(const v8bf*)&Ks[col][48 + 8 * hf]);
      v8f c = {};
      c = wmma_bf16(aq0, b0, c);
      c = wmma_bf16(aq1, b1, c);
      const float mval = mask[(size_t)bidx * S_ + n0 + col] * -1e9f;
      const size_t colg = n0 + col;
#pragma unroll
      for (int r = 0; r < 8; ++r) {
        const float v = c[r] * scale + mval;
        const int row = sbase + r + 8 * hf;
        attn[((size_t)bh * S_ + row) * S_ + colg] = v;
        const float nm = fmaxf(m[r], v);           // per-lane online softmax
        l[r] = l[r] * __expf(m[r] - nm) + __expf(v - nm);
        m[r] = nm;
      }
    }
    __syncthreads();
  }
  // merge 16 lanes (the 16 columns each row is striped over)
#pragma unroll
  for (int r = 0; r < 8; ++r) {
    float mm = m[r], ll = l[r];
#pragma unroll
    for (int off = 1; off < 16; off <<= 1) {
      const float mo = __shfl_xor(mm, off, 32);
      const float lo = __shfl_xor(ll, off, 32);
      const float nm = fmaxf(mm, mo);
      ll = ll * __expf(mm - nm) + lo * __expf(mo - nm);
      mm = nm;
    }
    if (l16 == r) {
      const int row = sbase + r + 8 * hf;
      const size_t idx = ((size_t)bh * S_ + row) * 2;
      rowstats[idx] = mm; rowstats[idx + 1] = ll;
    }
  }
}

// ---------------------------------------------------------------------------
// Kernel 3: attn = exp(logit - m)/l (written back, required output) and
// ctx = attn @ V, accumulated over 32-col K chunks. V staged via async copy.
// ---------------------------------------------------------------------------
__global__ __launch_bounds__(256)
void attn_ctx(float* __restrict__ attn, const float* __restrict__ rowstats,
              const bf16_t* __restrict__ VhT, float* __restrict__ ctx) {
  __shared__ bf16_t Ps[128][40];                   // probabilities [row][k]
  __shared__ bf16_t Vs[64][40];                    // V [d][k] (VhT is [d][s])
  __shared__ float  Ms[128], Li[128];
  const int t = threadIdx.x, lane = t & 31, wave = t >> 5;
  const int hf = lane >> 4, l16 = lane & 15;
  const int bh = blockIdx.y, bb = bh >> 4, hh = bh & 15;
  const int qbase = blockIdx.x * 128;

  if (t < 128) {
    const size_t idx = ((size_t)bh * S_ + qbase + t) * 2;
    const float ll = rowstats[idx + 1];
    Ms[t] = rowstats[idx];
    Li[t] = (ll > 0.f) ? (1.f / ll) : 0.f;
  }
  __syncthreads();

  v8f acc[4] = {};
  for (int kc = 0; kc < S_ / 32; ++kc) {
    const int k0 = kc * 32;
    {                                              // V tile: 64 d-rows x 32 s
      const int row = t >> 2, o8 = (t & 3) << 3;   // 256 x 8-elem chunks
      async_copy_b128(&VhT[((size_t)bh * DEPTH_ + row) * S_ + k0 + o8], &Vs[row][o8]);
    }
#pragma unroll
    for (int j = 0; j < 4; ++j) {                  // logits 128x32 -> probs
      const int lin = t + 256 * j;                 // 1024 float4
      const int row = lin >> 3, c4 = (lin & 7) << 2;
      const size_t g = ((size_t)bh * S_ + qbase + row) * S_ + k0 + c4;
      const float4 v = *(const float4*)&attn[g];
      const float mr = Ms[row], iv = Li[row];
      float4 p;
      p.x = __expf(v.x - mr) * iv; p.y = __expf(v.y - mr) * iv;
      p.z = __expf(v.z - mr) * iv; p.w = __expf(v.w - mr) * iv;
      *(float4*)&attn[g] = p;                      // normalized attn output
      Ps[row][c4 + 0] = (bf16_t)p.x; Ps[row][c4 + 1] = (bf16_t)p.y;
      Ps[row][c4 + 2] = (bf16_t)p.z; Ps[row][c4 + 3] = (bf16_t)p.w;
    }
    async_wait0();
    __syncthreads();

    const int lrow = wave * 16 + l16;
    const v16bf a = cat16(*(const v8bf*)&Ps[lrow][8 * hf],
                          *(const v8bf*)&Ps[lrow][16 + 8 * hf]);
#pragma unroll
    for (int t4 = 0; t4 < 4; ++t4) {
      const int col = t4 * 16 + l16;
      const v16bf b = cat16(*(const v8bf*)&Vs[col][8 * hf],
                            *(const v8bf*)&Vs[col][16 + 8 * hf]);
      acc[t4] = wmma_bf16(a, b, acc[t4]);
    }
    __syncthreads();
  }

#pragma unroll
  for (int t4 = 0; t4 < 4; ++t4) {                 // ctx [B][S][D] head-merged
    const int dcol = hh * DEPTH_ + t4 * 16 + l16;
#pragma unroll
    for (int r = 0; r < 8; ++r) {
      const int s = qbase + wave * 16 + r + 8 * hf;
      ctx[((size_t)bb * S_ + s) * D_ + dcol] = acc[t4][r];
    }
  }
}

// ---------------------------------------------------------------------------
extern "C" void kernel_launch(void* const* d_in, const int* in_sizes, int n_in,
                              void* d_out, int out_size, void* d_ws, size_t ws_size,
                              hipStream_t stream) {
  const float* q    = (const float*)d_in[0];
  const float* k    = (const float*)d_in[1];
  const float* v    = (const float*)d_in[2];
  const float* mask = (const float*)d_in[3];
  const float* Wq   = (const float*)d_in[4];
  const float* bq   = (const float*)d_in[5];
  const float* Wk   = (const float*)d_in[6];
  const float* bk   = (const float*)d_in[7];
  const float* Wv   = (const float*)d_in[8];
  const float* bv   = (const float*)d_in[9];
  const float* Wo   = (const float*)d_in[10];
  const float* bo   = (const float*)d_in[11];

  float* out  = (float*)d_out;
  float* attn = out + (size_t)B_ * S_ * D_;        // tuple output #2

  char* ws = (char*)d_ws;
  bf16_t* Qh       = (bf16_t*)(ws);                             //  8 MB
  bf16_t* Kh       = (bf16_t*)(ws + (size_t)8  * 1024 * 1024);  //  8 MB
  bf16_t* VhT      = (bf16_t*)(ws + (size_t)16 * 1024 * 1024);  //  8 MB
  float*  rowstats = (float*) (ws + (size_t)24 * 1024 * 1024);  //  0.5 MB
  float*  ctx      = (float*) (ws + (size_t)25 * 1024 * 1024);  // 16 MB

  const dim3 blk(256);
  const dim3 gGemm(D_ / 128, (B_ * S_) / 128);     // (8, 32)
  const dim3 gAttn(S_ / 128, BH_);                 // (16, 32)

  gemm_proj<0><<<gGemm, blk, 0, stream>>>(q, Wq, bq, (void*)Qh);
  gemm_proj<1><<<gGemm, blk, 0, stream>>>(k, Wk, bk, (void*)Kh);
  gemm_proj<2><<<gGemm, blk, 0, stream>>>(v, Wv, bv, (void*)VhT);
  attn_logits<<<gAttn, blk, 0, stream>>>(Qh, Kh, mask, attn, rowstats);
  attn_ctx  <<<gAttn, blk, 0, stream>>>(attn, rowstats, VhT, ctx);
  gemm_proj<3><<<gGemm, blk, 0, stream>>>(ctx, Wo, bo, (void*)out);
}
